// GNN_80281528697023
// MI455X (gfx1250) — matrix operand order repaired and checked by reading, not compiled, not run on hardware
//
#include <hip/hip_runtime.h>
#include <hip/hip_bf16.h>
#include <math.h>

typedef __attribute__((ext_vector_type(16))) _Float16 v16h;
typedef __attribute__((ext_vector_type(8)))  _Float16 v8h;
typedef __attribute__((ext_vector_type(8)))  float    v8f;

#define WAVES 8            // 256-thread blocks = 8 wave32
#define MAX_IT 10

// Per-wave LDS fence: wave32 executes in lockstep, all LDS traffic here is
// wave-private, so waiting DScnt==0 (+compiler memory barrier) is sufficient.
__device__ __forceinline__ void lds_fence() {
    asm volatile("s_wait_dscnt 0" ::: "memory");
}

// Hardware tanh (V_TANH_F32 is a CDNA5 transcendental). Prefer the builtin so
// the scheduler knows the TRANS hazard; otherwise inline asm + explicit v_nop
// to satisfy the "1 independent op after trans before use" rule.
__device__ __forceinline__ float fast_tanh(float x) {
#if __has_builtin(__builtin_amdgcn_tanhf)
    return __builtin_amdgcn_tanhf(x);
#else
    float r;
    asm volatile("v_tanh_f32 %0, %1\n\tv_nop" : "=v"(r) : "v"(x));
    return r;
#endif
}

__global__ void gnn_zero_kernel(float* __restrict__ p, int n) {
    int i = blockIdx.x * blockDim.x + threadIdx.x;
    if (i < n) p[i] = 0.0f;
}

// One GNN message-passing iteration over all edges.
// x = [child_id, arc_feats(16), state[child](4)]  (21 values, K-padded to 32)
// h = tanh(x @ W1 + b1)   -> WMMA1 (16x16x32 f16, N padded 5->16)
// m = tanh(h @ W2 + b2)   -> WMMA2 (16x16x32 f16, N padded 4->16)
// state_out[parent] += m  -> f32 atomics
__global__ void __launch_bounds__(256)
gnn_edge_pass(const float* __restrict__ arc,      // [E,17]
              const float* __restrict__ W1,       // [21,5]
              const float* __restrict__ b1,       // [5]
              const float* __restrict__ W2,       // [5,4]
              const float* __restrict__ b2,       // [4]
              const int*   __restrict__ par,      // [E]
              const int*   __restrict__ chi,      // [E]
              const float* __restrict__ sin_,     // [N,4]
              float*       __restrict__ sout,     // [N,4]
              int ntiles)
{
    __shared__ alignas(16) _Float16 xs[WAVES][16][32]; // x staging, K-padded
    __shared__ alignas(16) _Float16 hs[WAVES][16][32]; // h staging, K-padded
    __shared__ alignas(16) float    cs[WAVES][16];     // child ids as f32

    const int lane = threadIdx.x & 31;
    const int w    = threadIdx.x >> 5;
    const int m    = lane & 15;       // edge-in-tile / row index
    const int part = lane >> 4;       // which K-half this lane owns
    const int n16  = lane & 15;       // N index in C-matrix layout

    // Zero the K-padding once (staging only ever rewrites x[0..20], h[0..4]).
    #pragma unroll
    for (int i = 0; i < 16; ++i) {
        xs[w][m][part * 16 + i] = (_Float16)0.0f;
        hs[w][m][part * 16 + i] = (_Float16)0.0f;
    }

    // --- Build B-matrices (weights) in WMMA B layout: lane column N = lane%16,
    // element i = K in table order (8*part+i, 16+8*part+i-8). Zero-padded beyond real dims. ---
    v16h B1; v16h B2m;
    #pragma unroll
    for (int i = 0; i < 16; ++i) { B1[i] = (_Float16)0.0f; B2m[i] = (_Float16)0.0f; }
    #pragma unroll
    for (int i = 0; i < 16; ++i) {
        const int k = (i < 8) ? (8 * part + i) : (16 + 8 * part + (i - 8));
        if (n16 < 5 && k < 21) B1[i] = (_Float16)W1[k * 5 + n16];
        if (n16 < 4 && k < 5) B2m[i] = (_Float16)W2[k * 4 + n16];
    }
    const float bias1 = (n16 < 5) ? b1[n16] : 0.0f;
    const float bias2 = (n16 < 4) ? b2[n16] : 0.0f;
    // child-id column handled in f32 (overflows f16): rank-1 add to C1.
    const float w1r0n = (n16 < 5) ? W1[n16] : 0.0f;

    const int gwave  = blockIdx.x * WAVES + w;
    const int nwaves = gridDim.x * WAVES;

    lds_fence();

    for (int tile = gwave; tile < ntiles; tile += nwaves) {
        const int ebase = tile * 16;
        const int e = ebase + m;

        // ---- stage x into LDS (split the 17 arc columns across the 2 half-waves) ----
        if (part == 0) {
            xs[w][m][0] = (_Float16)0.0f;                  // child-id col -> f32 path
            #pragma unroll
            for (int j = 1; j < 9; ++j) xs[w][m][j] = (_Float16)arc[(size_t)e * 17 + j];
        } else {
            #pragma unroll
            for (int j = 9; j < 17; ++j) xs[w][m][j] = (_Float16)arc[(size_t)e * 17 + j];
            const int c = chi[e];
            cs[w][m] = (float)c;
            const float4 st = *(const float4*)(sin_ + (size_t)c * 4);
            xs[w][m][17] = (_Float16)st.x;
            xs[w][m][18] = (_Float16)st.y;
            xs[w][m][19] = (_Float16)st.z;
            xs[w][m][20] = (_Float16)st.w;
        }
        lds_fence();

        // ---- assemble A1: lane m holds K 0-7 & 16-23 of row m; lane m+16 holds 8-15 & 24-31 ----
        v16h A;
        {
            const v8h* p0 = (const v8h*)&xs[w][m][part * 8];
            v8h lo = p0[0];            // K base+0..7
            v8h hi = p0[2];            // K base+16..23 (16 halves further)
            #pragma unroll
            for (int i = 0; i < 8; ++i) { A[i] = lo[i]; A[i + 8] = hi[i]; }
        }

        // child ids for rows M = 8*part .. 8*part+7, as two b128 LDS loads
        float cid[8];
        {
            const float4* cp = (const float4*)&cs[w][8 * part];
            const float4 c0 = cp[0], c1v = cp[1];
            cid[0] = c0.x;  cid[1] = c0.y;  cid[2] = c0.z;  cid[3] = c0.w;
            cid[4] = c1v.x; cid[5] = c1v.y; cid[6] = c1v.z; cid[7] = c1v.w;
        }

        v8f c1 = {};
        c1 = __builtin_amdgcn_wmma_f32_16x16x32_f16(false, A, false, B1,
                                                    (short)0, c1, false, false);

        // ---- bias + child-id rank-1 + tanh; stage h (transpose via LDS) ----
        float t[8];
        #pragma unroll
        for (int j = 0; j < 8; ++j)
            t[j] = fast_tanh(c1[j] + cid[j] * w1r0n + bias1);
        if (n16 < 5) {
            #pragma unroll
            for (int j = 0; j < 8; ++j)
                hs[w][j + 8 * part][n16] = (_Float16)t[j];
        }
        lds_fence();

        // ---- assemble A2 from h (K 5..31 are zero-padded) ----
        {
            const v8h* p0 = (const v8h*)&hs[w][m][part * 8];
            v8h lo = p0[0];
            v8h hi = p0[2];
            #pragma unroll
            for (int i = 0; i < 8; ++i) { A[i] = lo[i]; A[i + 8] = hi[i]; }
        }

        v8f c2 = {};
        c2 = __builtin_amdgcn_wmma_f32_16x16x32_f16(false, A, false, B2m,
                                                    (short)0, c2, false, false);

        // ---- tanh + scatter-add to parents (C layout: VGPR j -> row j+8*part, col n16) ----
        #pragma unroll
        for (int j = 0; j < 8; ++j) {
            if (n16 < 4) {
                const int e2 = ebase + j + 8 * part;
                const int p  = par[e2];
                const float v = fast_tanh(c2[j] + bias2);
                __hip_atomic_fetch_add(&sout[(size_t)p * 4 + n16], v,
                                       __ATOMIC_RELAXED, __HIP_MEMORY_SCOPE_AGENT);
            }
        }
        lds_fence();   // keep LDS reuse ordered across tiles
    }
}

// Node output head: out = softmax(tanh(state @ Wo1 + bo1) @ Wo2 + bo2)
__global__ void __launch_bounds__(256)
gnn_output_pass(const float* __restrict__ state,  // [N,4]
                const float* __restrict__ Wo1,    // [4,5]
                const float* __restrict__ bo1,    // [5]
                const float* __restrict__ Wo2,    // [5,16]
                const float* __restrict__ bo2,    // [16]
                float*       __restrict__ out,    // [N,16]
                int N)
{
    const int i = blockIdx.x * blockDim.x + threadIdx.x;
    if (i >= N) return;
    const float4 s = *(const float4*)(state + (size_t)i * 4);
    float ho[5];
    #pragma unroll
    for (int k = 0; k < 5; ++k) {
        float a = bo1[k];
        a += s.x * Wo1[0 * 5 + k];
        a += s.y * Wo1[1 * 5 + k];
        a += s.z * Wo1[2 * 5 + k];
        a += s.w * Wo1[3 * 5 + k];
        ho[k] = fast_tanh(a);
    }
    float logit[16];
    float mx = -1e30f;
    #pragma unroll
    for (int o = 0; o < 16; ++o) {
        float a = bo2[o];
        #pragma unroll
        for (int k = 0; k < 5; ++k) a += ho[k] * Wo2[k * 16 + o];
        logit[o] = a;
        mx = fmaxf(mx, a);
    }
    float sum = 0.0f;
    #pragma unroll
    for (int o = 0; o < 16; ++o) { logit[o] = __expf(logit[o] - mx); sum += logit[o]; }
    const float inv = 1.0f / sum;
    #pragma unroll
    for (int o = 0; o < 16; ++o) out[(size_t)i * 16 + o] = logit[o] * inv;
}

extern "C" void kernel_launch(void* const* d_in, const int* in_sizes, int n_in,
                              void* d_out, int out_size, void* d_ws, size_t ws_size,
                              hipStream_t stream) {
    const float* arc = (const float*)d_in[0];
    const float* W1  = (const float*)d_in[1];
    const float* b1  = (const float*)d_in[2];
    const float* W2  = (const float*)d_in[3];
    const float* b2  = (const float*)d_in[4];
    const float* Wo1 = (const float*)d_in[5];
    const float* bo1 = (const float*)d_in[6];
    const float* Wo2 = (const float*)d_in[7];
    const float* bo2 = (const float*)d_in[8];
    const int* par   = (const int*)d_in[9];
    const int* chi   = (const int*)d_in[10];

    const int E = in_sizes[9];          // n edges
    const int N = out_size / 16;        // n nodes (OUT=16 classes)
    const int ntiles = E / 16;          // E is a multiple of 16

    float* sA = (float*)d_ws;           // [N,4] state ping
    float* sB = sA + (size_t)N * 4;     // [N,4] state pong

    const int zthr = 256;
    const int zblk = (N * 4 + zthr - 1) / zthr;
    const int eblk = 800;               // 800*8 = 6400 waves grid-striding tiles
    const int oblk = (N + 255) / 256;

    gnn_zero_kernel<<<zblk, zthr, 0, stream>>>(sA, N * 4);

    float* cur = sA;
    float* nxt = sB;
    for (int it = 0; it < MAX_IT; ++it) {
        gnn_zero_kernel<<<zblk, zthr, 0, stream>>>(nxt, N * 4);
        gnn_edge_pass<<<eblk, 256, 0, stream>>>(arc, W1, b1, W2, b2, par, chi,
                                                cur, nxt, ntiles);
        float* tmp = cur; cur = nxt; nxt = tmp;
    }

    gnn_output_pass<<<oblk, 256, 0, stream>>>(cur, Wo1, bo1, Wo2, bo2,
                                              (float*)d_out, N);
}
